// TransformerEncoderLayer_8186207666828
// MI455X (gfx1250) — compile-verified, matrix operand order checked by reading, not run on hardware
//
#include <hip/hip_runtime.h>

// ---------------- problem constants ----------------
constexpr int B   = 16;
constexpr int S   = 512;
constexpr int D   = 512;
constexpr int H   = 8;
constexpr int DK  = 64;   // == DV
constexpr int DFF = 2048;
constexpr float NEGINF = -1e18f;

using u16 = unsigned short;

// ---------------- WMMA types ----------------
typedef __bf16 bf16x16 __attribute__((ext_vector_type(16)));
typedef __bf16 bf16x2  __attribute__((ext_vector_type(2)));
typedef float  v8f     __attribute__((ext_vector_type(8)));
typedef int    v4i     __attribute__((vector_size(16)));   // matches async-LDS builtin

union FragU {               // 32 bytes
    bf16x16 v;
    uint4   q[2];
    bf16x2  h[8];
    u16     u[16];
};
union Pack8 {               // 16 bytes
    uint4 q;
    u16   u[8];
};

// Packed fp32 -> bf16 conversion: native v_cvt_pk_bf16_f32 when present.
__device__ __forceinline__ bf16x2 cvt2(float a, float b) {
#if __has_builtin(__builtin_amdgcn_cvt_pk_bf16_f32)
    return __builtin_amdgcn_cvt_pk_bf16_f32(a, b);
#else
    bf16x2 r; r.x = (__bf16)a; r.y = (__bf16)b; return r;
#endif
}
__device__ __forceinline__ u16 f2bf1(float x) {
    bf16x2 t = cvt2(x, x);
    return __builtin_bit_cast(u16, t.x);
}

__device__ __forceinline__ v8f wmma_bf16(bf16x16 a, bf16x16 b, v8f c) {
    return __builtin_amdgcn_wmma_f32_16x16x32_bf16(
        false, a, false, b, (short)0, c, false, false);
}

// ---------------- CDNA5 async global -> LDS copy (16B per lane) -------
#if __has_builtin(__builtin_amdgcn_global_load_async_to_lds_b128)
#define USE_ASYNC_LDS 1
#else
#define USE_ASYNC_LDS 0
#endif

__device__ __forceinline__ void copy16B_to_lds(u16* lds_dst, const u16* gsrc) {
#if USE_ASYNC_LDS
    __builtin_amdgcn_global_load_async_to_lds_b128(
        (v4i*)gsrc, (v4i*)lds_dst, 0, 0);
#else
    *(uint4*)lds_dst = *(const uint4*)gsrc;
#endif
}
__device__ __forceinline__ void wait_async0() {
#if USE_ASYNC_LDS
 #if __has_builtin(__builtin_amdgcn_s_wait_asynccnt)
    __builtin_amdgcn_s_wait_asynccnt(0);
 #else
    asm volatile("s_wait_asynccnt 0" ::: "memory");
 #endif
#endif
}

// 16x32 bf16 fragment from bf16 global memory (row stride ld elements).
// CDNA5 16-bit operand layout: lanes 0-15 row=lane, K=0..7/16..23;
// lanes 16-31 same rows, K=8..15/24..31 -> two 16B loads per lane.
__device__ __forceinline__ bf16x16 frag_from_bf16_global(const u16* p0, int ld) {
    const int lane = threadIdx.x & 31;
    const u16* p = p0 + (size_t)(lane & 15) * ld + ((lane >> 4) << 3);
    FragU f;
    f.q[0] = *(const uint4*)(p);
    f.q[1] = *(const uint4*)(p + 16);
    return f.v;
}

// Same fragment from fp32 global, converted with packed cvt (attn probs).
__device__ __forceinline__ bf16x16 frag_from_f32_global(const float* p0, int ld) {
    const int lane = threadIdx.x & 31;
    const float* p = p0 + (size_t)(lane & 15) * ld + ((lane >> 4) << 3);
    const float4 x0 = *(const float4*)(p + 0);
    const float4 x1 = *(const float4*)(p + 4);
    const float4 x2 = *(const float4*)(p + 16);
    const float4 x3 = *(const float4*)(p + 20);
    FragU f;
    f.h[0] = cvt2(x0.x, x0.y); f.h[1] = cvt2(x0.z, x0.w);
    f.h[2] = cvt2(x1.x, x1.y); f.h[3] = cvt2(x1.z, x1.w);
    f.h[4] = cvt2(x2.x, x2.y); f.h[5] = cvt2(x2.z, x2.w);
    f.h[6] = cvt2(x3.x, x3.y); f.h[7] = cvt2(x3.z, x3.w);
    return f.v;
}

// Fragment from an LDS bf16 tile with row stride 32 elements.
__device__ __forceinline__ bf16x16 frag_from_lds(const u16* smem, int row0) {
    const int lane = threadIdx.x & 31;
    const u16* p = smem + (size_t)(row0 + (lane & 15)) * 32 + ((lane >> 4) << 3);
    FragU f;
    f.q[0] = *(const uint4*)(p);
    f.q[1] = *(const uint4*)(p + 16);
    return f.v;
}

__device__ __forceinline__ float wave_sum(float x) {
    for (int o = 16; o > 0; o >>= 1) x += __shfl_xor(x, o, 32);
    return x;
}
__device__ __forceinline__ float wave_max(float x) {
    for (int o = 16; o > 0; o >>= 1) x = fmaxf(x, __shfl_xor(x, o, 32));
    return x;
}

// =====================================================================
// One-shot fp32 -> bf16 tensor conversion (8 elements / thread).
// =====================================================================
__global__ __launch_bounds__(256) void f32_to_bf16_kernel(
    const float* __restrict__ x, u16* __restrict__ y, int n8)
{
    const int i = blockIdx.x * 256 + threadIdx.x;
    if (i >= n8) return;
    const float4 a = ((const float4*)x)[2 * i + 0];
    const float4 b = ((const float4*)x)[2 * i + 1];
    union { uint4 q; bf16x2 h[4]; } o;
    o.h[0] = cvt2(a.x, a.y); o.h[1] = cvt2(a.z, a.w);
    o.h[2] = cvt2(b.x, b.y); o.h[3] = cvt2(b.z, b.w);
    ((uint4*)y)[i] = o.q;
}

// =====================================================================
// GEMM: C[M,N] = A[M,K] * W[N,K]^T + bias,  A/W bf16, C fp32 or bf16.
// Block = 256 threads (8 waves), tile 128x128, BK=32, wave tile 32x64.
// Double-buffered LDS staged with async global->LDS DMA (ASYNCcnt),
// pipelined: tile k+1 copies overlap tile k's 8 WMMAs; 1 barrier/iter.
// =====================================================================
template<bool RELU, bool HASRES, bool OUTBF>
__global__ __launch_bounds__(256) void gemm_bf16_kernel(
    const u16* __restrict__ A, const u16* __restrict__ W,
    const float* __restrict__ bias, const float* __restrict__ resid,
    void* __restrict__ Cout, int M, int N, int K)
{
    __shared__ __align__(16) u16 As[2][128 * 32];
    __shared__ __align__(16) u16 Bs[2][128 * 32];

    const int t     = threadIdx.x;
    const int lane  = t & 31;
    const int wid   = t >> 5;
    const int waveM = wid >> 1;            // 0..3  -> 32 rows each
    const int waveN = wid & 1;             // 0..1  -> 64 cols each
    const int blockM = blockIdx.y * 128;
    const int blockN = blockIdx.x * 128;

    const int row = t >> 1;
    const int kq  = (t & 1) * 16;
    const u16* gA = A + (size_t)(blockM + row) * K + kq;
    const u16* gB = W + (size_t)(blockN + row) * K + kq;
    u16* const lA = &As[0][0] + row * 32 + kq;
    u16* const lB = &Bs[0][0] + row * 32 + kq;
    const int bufStride = 128 * 32;

    auto stage = [&](int buf, int kb) {
        copy16B_to_lds(lA + buf * bufStride,     gA + kb);
        copy16B_to_lds(lA + buf * bufStride + 8, gA + kb + 8);
        copy16B_to_lds(lB + buf * bufStride,     gB + kb);
        copy16B_to_lds(lB + buf * bufStride + 8, gB + kb + 8);
    };

    v8f acc[2][4] = {};
    stage(0, 0);
    int cur = 0;
    for (int kb = 0; kb < K; kb += 32) {
        wait_async0();
        __syncthreads();
        if (kb + 32 < K) stage(cur ^ 1, kb + 32);   // overlap with compute

        const u16* Ac = &As[cur][0];
        const u16* Bc = &Bs[cur][0];
        const bf16x16 a0 = frag_from_lds(Ac, waveM * 32 + 0);
        const bf16x16 a1 = frag_from_lds(Ac, waveM * 32 + 16);
        #pragma unroll
        for (int tn = 0; tn < 4; ++tn) {
            const bf16x16 bb = frag_from_lds(Bc, waveN * 64 + tn * 16);
            acc[0][tn] = wmma_bf16(a0, bb, acc[0][tn]);
            acc[1][tn] = wmma_bf16(a1, bb, acc[1][tn]);
        }
        cur ^= 1;
    }

    const int lh = lane >> 4, ln = lane & 15;
    #pragma unroll
    for (int tm = 0; tm < 2; ++tm)
    #pragma unroll
    for (int tn = 0; tn < 4; ++tn) {
        const int m0 = blockM + waveM * 32 + tm * 16 + lh * 8;
        const int n  = blockN + waveN * 64 + tn * 16 + ln;
        const float bv = bias[n];
        const v8f c = acc[tm][tn];
        #pragma unroll
        for (int r = 0; r < 8; ++r) {
            float x = c[r] + bv;
            if (RELU)   x = fmaxf(x, 0.0f);
            if (HASRES) x += resid[(size_t)(m0 + r) * N + n];
            const size_t idx = (size_t)(m0 + r) * N + n;
            if constexpr (OUTBF) ((u16*)Cout)[idx] = f2bf1(x);
            else                 ((float*)Cout)[idx] = x;
        }
    }
}

// =====================================================================
// scores[b,h,q,k] = (Q.K^T)/8 with structural-head gating + pad mask.
// Q,K are bf16 [B*S, H*DK].  Block tile 64q x 128k, wave tile 32x32.
// =====================================================================
__global__ __launch_bounds__(256) void attn_scores_kernel(
    const u16* __restrict__ Q, const u16* __restrict__ Kmat,
    const int* __restrict__ c0, const int* __restrict__ c1,
    const int* __restrict__ c2, const int* __restrict__ c3,
    const int* __restrict__ c4,
    const int* __restrict__ mask, const int* __restrict__ heads_type,
    float* __restrict__ scores)
{
    const int t = threadIdx.x, lane = t & 31, wid = t >> 5;
    const int waveQ = wid >> 2;           // 0..1
    const int waveK = wid & 3;            // 0..3
    const int bh = blockIdx.z, b = bh >> 3, h = bh & 7;
    const int q0 = blockIdx.y * 64 + waveQ * 32;
    const int k0 = blockIdx.x * 128 + waveK * 32;

    const u16* Qp = Q    + (size_t)(b * S) * D + h * DK;
    const u16* Kp = Kmat + (size_t)(b * S) * D + h * DK;

    v8f acc[2][2] = {};
    #pragma unroll
    for (int kk = 0; kk < DK; kk += 32) {
        const bf16x16 a0 = frag_from_bf16_global(Qp + (size_t)(q0 + 0 ) * D + kk, D);
        const bf16x16 a1 = frag_from_bf16_global(Qp + (size_t)(q0 + 16) * D + kk, D);
        const bf16x16 b0 = frag_from_bf16_global(Kp + (size_t)(k0 + 0 ) * D + kk, D);
        const bf16x16 b1 = frag_from_bf16_global(Kp + (size_t)(k0 + 16) * D + kk, D);
        acc[0][0] = wmma_bf16(a0, b0, acc[0][0]);
        acc[0][1] = wmma_bf16(a0, b1, acc[0][1]);
        acc[1][0] = wmma_bf16(a1, b0, acc[1][0]);
        acc[1][1] = wmma_bf16(a1, b1, acc[1][1]);
    }

    const int ht = heads_type[0];
    const int* cp = nullptr;
    if (ht) {
        if      (h == 0) cp = c0;
        else if (h == 1) cp = c1;
        else if (h == 2) cp = c2;
        else if (h == 3) cp = c3;
        else if (h == 4) cp = c4;
    }

    const int lh = lane >> 4, ln = lane & 15;
    float* sOut = scores + (size_t)bh * S * S;
    #pragma unroll
    for (int tm = 0; tm < 2; ++tm)
    #pragma unroll
    for (int tn = 0; tn < 2; ++tn) {
        const int qq0  = q0 + tm * 16 + lh * 8;
        const int kcol = k0 + tn * 16 + ln;
        const v8f c = acc[tm][tn];
        #pragma unroll
        for (int r = 0; r < 8; ++r) {
            const int qi = qq0 + r;
            float sv = c[r] * 0.125f;                      // 1/sqrt(64)
            const size_t mi = (size_t)(b * S + qi) * S + kcol;
            if (cp && cp[mi] <= 0) sv = NEGINF;
            if (mask[mi] != 0)     sv = NEGINF;
            sOut[(size_t)qi * S + kcol] = sv;
        }
    }
}

// ============== softmax over last dim (512), one wave per row =========
__global__ __launch_bounds__(256) void softmax_kernel(float* __restrict__ attn)
{
    const int wid = threadIdx.x >> 5, lane = threadIdx.x & 31;
    const size_t row = (size_t)blockIdx.x * 8 + wid;
    float* p = attn + row * S;
    float v[16];
    float m = -3.0e38f;
    #pragma unroll
    for (int j = 0; j < 16; ++j) { v[j] = p[j * 32 + lane]; m = fmaxf(m, v[j]); }
    m = wave_max(m);
    float sum = 0.f;
    #pragma unroll
    for (int j = 0; j < 16; ++j) { v[j] = __expf(v[j] - m); sum += v[j]; }
    sum = wave_sum(sum);
    const float inv = 1.0f / sum;
    #pragma unroll
    for (int j = 0; j < 16; ++j) p[j * 32 + lane] = v[j] * inv;
}

// =====================================================================
// ctx[b,q,h*64+d] = attn[b,h,q,:] @ V[b,:,h*64+d]; V bf16, ctx bf16.
// Block: 128 q, 8 waves (4x2), V tile transposed into LDS.
// =====================================================================
__global__ __launch_bounds__(256) void attn_v_kernel(
    const float* __restrict__ attn, const u16* __restrict__ V,
    u16* __restrict__ ctx)
{
    __shared__ __align__(16) u16 Vs[64 * 32];   // [d][s]
    const int t = threadIdx.x, lane = t & 31, wid = t >> 5;
    const int waveQ = wid >> 1;           // 0..3
    const int waveD = wid & 1;            // 0..1
    const int bh = blockIdx.y, b = bh >> 3, h = bh & 7;
    const int q0 = blockIdx.x * 128 + waveQ * 32;

    const float* Ap = attn + (size_t)bh * S * S;
    const u16*   Vp = V + (size_t)(b * S) * D + h * DK;

    v8f acc[2][2] = {};
    for (int s0 = 0; s0 < S; s0 += 32) {
        { // stage V[s0..s0+32, 0..64] transposed -> Vs[d][s]
            const int s  = t >> 3;
            const int dq = (t & 7) * 8;
            Pack8 x;
            x.q = *(const uint4*)(Vp + (size_t)(s0 + s) * D + dq);
            #pragma unroll
            for (int j = 0; j < 8; ++j) Vs[(dq + j) * 32 + s] = x.u[j];
        }
        __syncthreads();
        const bf16x16 a0 = frag_from_f32_global(Ap + (size_t)(q0 + 0 ) * S + s0, S);
        const bf16x16 a1 = frag_from_f32_global(Ap + (size_t)(q0 + 16) * S + s0, S);
        const bf16x16 b0 = frag_from_lds(Vs, waveD * 32 + 0);
        const bf16x16 b1 = frag_from_lds(Vs, waveD * 32 + 16);
        acc[0][0] = wmma_bf16(a0, b0, acc[0][0]);
        acc[0][1] = wmma_bf16(a0, b1, acc[0][1]);
        acc[1][0] = wmma_bf16(a1, b0, acc[1][0]);
        acc[1][1] = wmma_bf16(a1, b1, acc[1][1]);
        __syncthreads();
    }

    const int lh = lane >> 4, ln = lane & 15;
    #pragma unroll
    for (int tm = 0; tm < 2; ++tm)
    #pragma unroll
    for (int tn = 0; tn < 2; ++tn) {
        const int qq0 = q0 + tm * 16 + lh * 8;
        const int d   = waveD * 32 + tn * 16 + ln;
        const v8f c = acc[tm][tn];
        #pragma unroll
        for (int r = 0; r < 8; ++r)
            ctx[(size_t)(b * S + qq0 + r) * D + h * DK + d] = f2bf1(c[r]);
    }
}

// ====== LayerNorm (opt residual add), wave per row, fp32/bf16 out =====
template<bool RES, bool OUTBF>
__global__ __launch_bounds__(256) void layernorm_kernel(
    const float* __restrict__ x, const float* __restrict__ resid,
    const float* __restrict__ g, const float* __restrict__ beta,
    void* __restrict__ out)
{
    const int wid = threadIdx.x >> 5, lane = threadIdx.x & 31;
    const size_t row = (size_t)blockIdx.x * 8 + wid;
    const float* p = x + row * D;
    float v[16];
    float s = 0.f;
    #pragma unroll
    for (int j = 0; j < 16; ++j) {
        float val = p[j * 32 + lane];
        if (RES) val += resid[row * D + j * 32 + lane];
        v[j] = val; s += val;
    }
    const float mean = wave_sum(s) * (1.0f / 512.0f);
    float s2 = 0.f;
    #pragma unroll
    for (int j = 0; j < 16; ++j) { const float d = v[j] - mean; s2 += d * d; }
    const float rstd = rsqrtf(wave_sum(s2) * (1.0f / 512.0f) + 1e-6f);
    #pragma unroll
    for (int j = 0; j < 16; ++j) {
        const int c = j * 32 + lane;
        const float y = (v[j] - mean) * rstd * g[c] + beta[c];
        if constexpr (OUTBF) ((u16*)out)[row * D + c] = f2bf1(y);
        else                 ((float*)out)[row * D + c] = y;
    }
}

// =====================================================================
extern "C" void kernel_launch(void* const* d_in, const int* in_sizes, int n_in,
                              void* d_out, int out_size, void* d_ws, size_t ws_size,
                              hipStream_t stream)
{
    (void)in_sizes; (void)n_in; (void)out_size; (void)ws_size;

    const float* inputs = (const float*)d_in[0];
    const int*   mask   = (const int*)d_in[1];
    const int*   c0     = (const int*)d_in[2];
    const int*   c1     = (const int*)d_in[3];
    const int*   c2     = (const int*)d_in[4];
    const int*   c3     = (const int*)d_in[5];
    const int*   c4     = (const int*)d_in[6];
    const int*   ht     = (const int*)d_in[7];
    const float* Wq = (const float*)d_in[8],  *bq = (const float*)d_in[9];
    const float* Wk = (const float*)d_in[10], *bk = (const float*)d_in[11];
    const float* Wv = (const float*)d_in[12], *bv = (const float*)d_in[13];
    const float* Wo = (const float*)d_in[14], *bo = (const float*)d_in[15];
    const float* ln1g = (const float*)d_in[16], *ln1b = (const float*)d_in[17];
    const float* W1 = (const float*)d_in[18], *b1 = (const float*)d_in[19];
    const float* W2 = (const float*)d_in[20], *b2 = (const float*)d_in[21];
    const float* ln2g = (const float*)d_in[22], *ln2b = (const float*)d_in[23];

    float* out     = (float*)d_out;
    float* ffn_out = out;                                // [B*S*D]
    float* attn    = out + (size_t)B * S * D;            // [B*H*S*S]

    const int M = B * S;                                 // 8192

    // ---------------- workspace layout (bytes), total ~118 MiB --------
    char* w = (char*)d_ws;
    const size_t ACT_H  = (size_t)M * D * 2;             // 8 MiB (bf16 act)
    const size_t WSQ_H  = (size_t)D * D * 2;             // 512 KiB
    const size_t WFF_H  = (size_t)DFF * D * 2;           // 2 MiB
    const size_t ACT_F  = (size_t)M * D * 4;             // 16 MiB (fp32 act)
    const size_t INT_H  = (size_t)M * DFF * 2;           // 32 MiB

    u16*   in_h   = (u16*)w;  w += ACT_H;
    u16*   Wq_h   = (u16*)w;  w += WSQ_H;
    u16*   Wk_h   = (u16*)w;  w += WSQ_H;
    u16*   Wv_h   = (u16*)w;  w += WSQ_H;
    u16*   Wo_h   = (u16*)w;  w += WSQ_H;
    u16*   W1_h   = (u16*)w;  w += WFF_H;
    u16*   W2_h   = (u16*)w;  w += WFF_H;
    u16*   Qh     = (u16*)w;  w += ACT_H;
    u16*   Kh     = (u16*)w;  w += ACT_H;
    u16*   Vh     = (u16*)w;  w += ACT_H;
    u16*   ctx_h  = (u16*)w;  w += ACT_H;
    float* ctxf   = (float*)w; w += ACT_F;               // Wo output (context)
    float* out1   = (float*)w; w += ACT_F;               // LN1 output
    u16*   ln2_h  = (u16*)w;  w += ACT_H;                // LN2 output
    u16*   inter_h= (u16*)w;  w += INT_H;                // FFN1 output

    const dim3 blk(256);
    const dim3 gD  (D   / 128, M / 128);                 // N=512 GEMMs
    const dim3 gFF (DFF / 128, M / 128);                 // N=2048 GEMM

    // 0) one-shot fp32 -> bf16 conversions
    auto cvt = [&](const float* src, u16* dst, size_t n) {
        const int n8 = (int)(n / 8);
        f32_to_bf16_kernel<<<dim3((n8 + 255) / 256), blk, 0, stream>>>(src, dst, n8);
    };
    cvt(inputs, in_h, (size_t)M * D);
    cvt(Wq, Wq_h, (size_t)D * D);
    cvt(Wk, Wk_h, (size_t)D * D);
    cvt(Wv, Wv_h, (size_t)D * D);
    cvt(Wo, Wo_h, (size_t)D * D);
    cvt(W1, W1_h, (size_t)DFF * D);
    cvt(W2, W2_h, (size_t)DFF * D);

    // 1) QKV projections -> bf16
    gemm_bf16_kernel<false,false,true ><<<gD, blk, 0, stream>>>(in_h, Wq_h, bq, nullptr, Qh, M, D, D);
    gemm_bf16_kernel<false,false,true ><<<gD, blk, 0, stream>>>(in_h, Wk_h, bk, nullptr, Kh, M, D, D);
    gemm_bf16_kernel<false,false,true ><<<gD, blk, 0, stream>>>(in_h, Wv_h, bv, nullptr, Vh, M, D, D);

    // 2) scores (+ structural gating + pad mask) -> attn region of d_out
    attn_scores_kernel<<<dim3(S / 128, S / 64, B * H), blk, 0, stream>>>(
        Qh, Kh, c0, c1, c2, c3, c4, mask, ht, attn);

    // 3) softmax in place (fp32 output required)
    softmax_kernel<<<dim3((B * H * S) / 8), blk, 0, stream>>>(attn);

    // 4) attn @ V -> ctx (bf16)
    attn_v_kernel<<<dim3(S / 128, B * H), blk, 0, stream>>>(attn, Vh, ctx_h);

    // 5) output projection -> context (fp32, feeds residual LN)
    gemm_bf16_kernel<false,false,false><<<gD, blk, 0, stream>>>(ctx_h, Wo_h, bo, nullptr, ctxf, M, D, D);

    // 6) residual + LN1 -> out1 (fp32, reused as FFN2 residual)
    layernorm_kernel<true ,false><<<dim3(M / 8), blk, 0, stream>>>(ctxf, inputs, ln1g, ln1b, out1);

    // 7) LN2 -> bf16
    layernorm_kernel<false,true ><<<dim3(M / 8), blk, 0, stream>>>(out1, nullptr, ln2g, ln2b, ln2_h);

    // 8) FFN1 (fused ReLU) -> inter (bf16)
    gemm_bf16_kernel<true ,false,true ><<<gFF, blk, 0, stream>>>(ln2_h, W1_h, b1, nullptr, inter_h, M, DFF, D);

    // 9) FFN2 (+ residual out1) -> ffn_out region of d_out (fp32)
    gemm_bf16_kernel<false,true ,false><<<gD, blk, 0, stream>>>(inter_h, W2_h, b2, out1, ffn_out, M, D, DFF);
}
